// MultiLangMap_22686017258037
// MI455X (gfx1250) — matrix operand order repaired and checked by reading, not compiled, not run on hardware
//
#include <hip/hip_runtime.h>
#include <hip/hip_bf16.h>
#include <math.h>

typedef __attribute__((ext_vector_type(16))) _Float16 v16h;
typedef __attribute__((ext_vector_type(8)))  _Float16 v8h;
typedef __attribute__((ext_vector_type(8)))  float    v8f;
typedef __attribute__((ext_vector_type(4)))  int      v4i;

#if __has_builtin(__builtin_amdgcn_global_load_async_to_lds_b128)
#define ATHENA_ASYNC_LDS 1
#endif
#if __has_builtin(__builtin_amdgcn_s_wait_asynccnt)
#define ATHENA_WAIT_ASYNC_BUILTIN 1
#endif

namespace {

constexpr int kB  = 16;
constexpr int kT  = 512;
constexpr int kD  = 256;
constexpr int kH  = 4;
constexpr int kDK = 64;
constexpr int kFF = 1024;
constexpr int kN  = 10;
constexpr int kKC = 128;  // GEMM K-chunk staged in LDS (halves)

union AFrag { v16h v; v8h h[2]; };

typedef __attribute__((address_space(1))) v4i v4i_g;  // global
typedef __attribute__((address_space(3))) v4i v4i_l;  // LDS

__device__ __forceinline__ v8f wmma_f16(v16h a, v16h b, v8f c) {
  return __builtin_amdgcn_wmma_f32_16x16x32_f16(false, a, false, b, (short)0, c,
                                                false, false);
}

// A operand (16x32 f16): lane<16 -> row = l16, halves {kt..kt+7, kt+16..kt+23}
//                        lane>=16 -> same row, halves {kt+8..kt+15, kt+24..kt+31}
__device__ __forceinline__ v16h load_a(const _Float16* rowp, int kt, int half) {
  AFrag f;
  const _Float16* p = rowp + kt + half * 8;
  f.h[0] = *(const v8h*)p;
  f.h[1] = *(const v8h*)(p + 16);
  return f.v;
}

// B operand (32x16 f16): lane<16 -> col = l16, halves {kt..kt+7, kt+16..kt+23}
//                        lane>=16 -> same col, halves {kt+8..kt+15, kt+24..kt+31}
__device__ __forceinline__ v16h load_b(const _Float16* colp, int kt, int half) {
  AFrag f;
  const _Float16* p = colp + kt + half * 8;
  f.h[0] = *(const v8h*)p;
  f.h[1] = *(const v8h*)(p + 16);
  return f.v;
}

// 16-byte global -> LDS copy; async (ASYNCcnt) when available
__device__ __forceinline__ void copy16_to_lds(const _Float16* g, _Float16* l) {
#ifdef ATHENA_ASYNC_LDS
  __builtin_amdgcn_global_load_async_to_lds_b128((v4i_g*)g, (v4i_l*)l, 0, 0);
#else
  *(v8h*)l = *(const v8h*)g;
#endif
}

__device__ __forceinline__ void wait_async_copies() {
#ifdef ATHENA_ASYNC_LDS
#ifdef ATHENA_WAIT_ASYNC_BUILTIN
  __builtin_amdgcn_s_wait_asynccnt(0);
#else
  asm volatile("s_wait_asynccnt 0" ::: "memory");
#endif
#endif
}

// ---------------- weight convert + transpose (f32 [z][k][n] -> f16 [z][n][k])
__global__ __launch_bounds__(256) void cvt_transpose_kernel(
    const float* __restrict__ in, _Float16* __restrict__ out, int K, int Nc,
    int total) {
  for (int i = blockIdx.x * 256 + threadIdx.x; i < total;
       i += gridDim.x * 256) {
    int k = i % K;
    int n = (i / K) % Nc;
    int z = i / (K * Nc);
    out[i] = (_Float16)in[((size_t)z * K + k) * Nc + n];
  }
}

// ---------------- h0 = x*sqrt(D) + positional encoding
__global__ __launch_bounds__(256) void prep_kernel(const float* __restrict__ x,
                                                   float* __restrict__ h0) {
  int i = blockIdx.x * 256 + threadIdx.x;
  if (i >= kB * kT * kD) return;
  int d = i & (kD - 1);
  int t = (i >> 8) & (kT - 1);
  float i2 = (float)(d & ~1);
  float div = __expf(i2 * (-9.210340371976184f / (float)kD));  // ln(1e4)
  float ang = (float)t * div;
  float pe = (d & 1) ? __cosf(ang) : __sinf(ang);
  h0[i] = x[i] * 16.0f + pe;  // sqrt(256) = 16
}

// ---------------- LayerNorm, per-z gamma/beta, output f16 or f32
template <bool F16OUT>
__global__ __launch_bounds__(256) void ln_kernel(
    const float* __restrict__ in, const float* __restrict__ g,
    const float* __restrict__ be, const int* __restrict__ lids,
    void* __restrict__ out) {
  __shared__ float red[256];
  const int row = blockIdx.x;   // b*T + t
  const int b = row >> 9;       // /512
  const int z = lids[b];
  const int t = threadIdx.x;
  float v = in[(size_t)row * kD + t];
  red[t] = v;
  __syncthreads();
  for (int s = 128; s > 0; s >>= 1) {
    if (t < s) red[t] += red[t + s];
    __syncthreads();
  }
  float mean = red[0] * (1.0f / kD);
  __syncthreads();
  float dv = v - mean;
  red[t] = dv * dv;
  __syncthreads();
  for (int s = 128; s > 0; s >>= 1) {
    if (t < s) red[t] += red[t + s];
    __syncthreads();
  }
  float var = red[0] * (1.0f / kD);
  float nv = dv * rsqrtf(var + 1e-12f) * g[(size_t)z * kD + t] +
             be[(size_t)z * kD + t];
  if (F16OUT)
    ((_Float16*)out)[(size_t)row * kD + t] = (_Float16)nv;
  else
    ((float*)out)[(size_t)row * kD + t] = nv;
}

// ---------------- WMMA GEMM: 4 waves/block, 64x64 tile, B panel staged in LDS
constexpr int STORE_F16 = 0;      // f16 row-major [b][t][NC]
constexpr int STORE_VT = 1;       // f16 V-transposed [b][h][dk][t]
constexpr int STORE_F32RES = 2;   // f32 row-major, += RES
constexpr int STORE_F16RELU = 3;  // f16 row-major, relu

template <int MODE>
__global__ __launch_bounds__(128) void gemm_kernel(
    const _Float16* __restrict__ A,   // [B][T][K] f16
    const _Float16* __restrict__ WT,  // [N][NC][K] f16 (pre-transposed)
    const float* __restrict__ bias,   // [N][NC] f32
    const float* __restrict__ RES,    // [B][T][NC] f32 or nullptr
    void* __restrict__ OUT, const int* __restrict__ lids, int K, int NC) {
  __shared__ _Float16 panel[2][64 * kKC];  // 32 KB double-buffered B panel
  const int tb = blockIdx.x;  // 64-row block
  const int tn = blockIdx.y;  // 64-col group
  const int b = blockIdx.z;
  const int z = lids[b];
  const int tid = threadIdx.x;
  const int wave = tid >> 5;
  const int lane = tid & 31;
  const int half = lane >> 4, l16 = lane & 15;
  const int tm = tb * 4 + wave;  // this wave's 16-row tile

  const _Float16* Ab = A + ((size_t)b * kT + tm * 16 + l16) * K;
  const _Float16* Wz = WT + ((size_t)z * NC + tn * 64) * K;

  const int nchunk = K / kKC;
  // one chunk = 64 cols x kKC halves = 1024 x 16B units; 8 units per thread
  auto stage = [&](int c, int buf) {
#pragma unroll
    for (int i = 0; i < 8; i++) {
      int u = tid + i * 128;
      int col = u >> 4;
      int k8 = (u & 15) * 8;
      copy16_to_lds(Wz + (size_t)col * K + c * kKC + k8,
                    &panel[buf][col * kKC + k8]);
    }
  };

  stage(0, 0);
  wait_async_copies();
  __syncthreads();

  v8f acc[4] = {};
  for (int c = 0; c < nchunk; ++c) {
    if (c + 1 < nchunk) stage(c + 1, (c + 1) & 1);  // overlap copy w/ compute
    const _Float16* pb = panel[c & 1];
#pragma unroll
    for (int kt = 0; kt < kKC; kt += 32) {
      v16h af = load_a(Ab, c * kKC + kt, half);
#pragma unroll
      for (int j = 0; j < 4; j++) {
        v16h bf = load_b(pb + (j * 16 + l16) * kKC, kt, half);
        acc[j] = wmma_f16(af, bf, acc[j]);
      }
    }
    if (c + 1 < nchunk) {
      wait_async_copies();
      __syncthreads();
    }
  }

#pragma unroll
  for (int j = 0; j < 4; j++) {
    const int col = tn * 64 + j * 16 + l16;
    const float bv = bias[(size_t)z * NC + col];
    if (MODE == STORE_VT) {
      const int hh = col >> 6, dk = col & 63;
      v8h pk;
#pragma unroll
      for (int r = 0; r < 8; r++) pk[r] = (_Float16)(acc[j][r] + bv);
      _Float16* p = (_Float16*)OUT +
                    (((size_t)b * kH + hh) * kDK + dk) * kT + tm * 16 + half * 8;
      *(v8h*)p = pk;
    } else {
#pragma unroll
      for (int r = 0; r < 8; r++) {
        const int m = tm * 16 + half * 8 + r;
        float val = acc[j][r] + bv;
        if (MODE == STORE_F16RELU) val = fmaxf(val, 0.0f);
        const size_t idx = ((size_t)b * kT + m) * NC + col;
        if (MODE == STORE_F32RES)
          ((float*)OUT)[idx] = val + RES[idx];
        else
          ((_Float16*)OUT)[idx] = (_Float16)val;
      }
    }
  }
}

// ---------------- attention: per (b, head, 16-row q tile), one wave
__global__ __launch_bounds__(32) void attn_kernel(
    const _Float16* __restrict__ Q16,  // [B][T][D]
    const _Float16* __restrict__ K16,  // [B][T][D]
    const _Float16* __restrict__ VT,   // [B][H][DK][T]
    _Float16* __restrict__ CTX,        // [B][T][D]
    const int* __restrict__ lens) {
  __shared__ float sc[16 * kT];  // 32 KB score strip
  const int tq = blockIdx.x, hh = blockIdx.y, b = blockIdx.z;
  const int lane = threadIdx.x, half = lane >> 4, l16 = lane & 15;
  const int len = lens[b];

  const _Float16* Qb =
      Q16 + ((size_t)b * kT + tq * 16 + l16) * kD + hh * kDK;
  const _Float16* Kb = K16 + (size_t)b * kT * kD + hh * kDK;

  const v16h qa0 = load_a(Qb, 0, half);
  const v16h qa1 = load_a(Qb, 32, half);

  // scores = (Q Kt) * 1/sqrt(64), masked to key < len
  for (int tk = 0; tk < kT / 16; ++tk) {
    const _Float16* kp = Kb + (size_t)(tk * 16 + l16) * kD;
    v16h kb0 = load_b(kp, 0, half);
    v16h kb1 = load_b(kp, 32, half);
    v8f s = {};
    s = wmma_f16(qa0, kb0, s);
    s = wmma_f16(qa1, kb1, s);
#pragma unroll
    for (int r = 0; r < 8; r++) {
      const int row = half * 8 + r;
      const int col = tk * 16 + l16;
      sc[row * kT + col] = (col < len) ? s[r] * 0.125f : -3.0e38f;
    }
  }
  __syncthreads();

  // masked softmax over each of the 16 rows (wave32 shuffle reductions)
  for (int m = 0; m < 16; m++) {
    float mx = -3.0e38f;
    for (int c = lane; c < kT; c += 32) mx = fmaxf(mx, sc[m * kT + c]);
    for (int o = 16; o; o >>= 1) mx = fmaxf(mx, __shfl_xor(mx, o, 32));
    float sum = 0.0f;
    for (int c = lane; c < kT; c += 32) {
      float v = sc[m * kT + c];
      float e = (v > -1.0e38f) ? __expf(v - mx) : 0.0f;
      sc[m * kT + c] = e;
      sum += e;
    }
    for (int o = 16; o; o >>= 1) sum += __shfl_xor(sum, o, 32);
    const float inv = 1.0f / sum;
    for (int c = lane; c < kT; c += 32) sc[m * kT + c] *= inv;
  }
  __syncthreads();

  // ctx(16x64) = P(16x512) @ V(512x64), B operand from V-transposed layout
  const _Float16* Vb = VT + ((size_t)b * kH + hh) * kDK * kT;
  v8f acc[4] = {};
  for (int kt = 0; kt < kT; kt += 32) {
    AFrag pa;
    const float* pr = &sc[l16 * kT + kt + half * 8];
#pragma unroll
    for (int e = 0; e < 8; e++) {
      pa.v[e] = (_Float16)pr[e];
      pa.v[8 + e] = (_Float16)pr[16 + e];
    }
#pragma unroll
    for (int j = 0; j < 4; j++) {
      const _Float16* vp = Vb + (size_t)(j * 16 + l16) * kT;
      v16h bf = load_b(vp, kt, half);
      acc[j] = wmma_f16(pa.v, bf, acc[j]);
    }
  }
#pragma unroll
  for (int j = 0; j < 4; j++) {
#pragma unroll
    for (int r = 0; r < 8; r++) {
      const int t = tq * 16 + half * 8 + r;
      const int dk = j * 16 + l16;
      CTX[((size_t)b * kT + t) * kD + hh * kDK + dk] = (_Float16)acc[j][r];
    }
  }
}

}  // namespace

extern "C" void kernel_launch(void* const* d_in, const int* in_sizes, int n_in,
                              void* d_out, int out_size, void* d_ws,
                              size_t ws_size, hipStream_t stream) {
  (void)in_sizes; (void)n_in; (void)out_size; (void)ws_size;
  const float* x   = (const float*)d_in[0];
  const int* xlen  = (const int*)d_in[1];
  const int* lids  = (const int*)d_in[2];
  const float* Wq  = (const float*)d_in[3];
  const float* bq  = (const float*)d_in[4];
  const float* Wk  = (const float*)d_in[5];
  const float* bk  = (const float*)d_in[6];
  const float* Wv  = (const float*)d_in[7];
  const float* bv  = (const float*)d_in[8];
  const float* Wo  = (const float*)d_in[9];
  const float* bo  = (const float*)d_in[10];
  const float* W1  = (const float*)d_in[11];
  const float* b1  = (const float*)d_in[12];
  const float* W2  = (const float*)d_in[13];
  const float* b2  = (const float*)d_in[14];
  const float* g1  = (const float*)d_in[15];
  const float* be1 = (const float*)d_in[16];
  const float* g2  = (const float*)d_in[17];
  const float* be2 = (const float*)d_in[18];
  const float* gf  = (const float*)d_in[19];
  const float* bef = (const float*)d_in[20];

  char* ws = (char*)d_ws;
  size_t off = 0;
  auto alloc = [&](size_t bytes) -> void* {
    void* p = ws + off;
    off += (bytes + 255) & ~(size_t)255;
    return p;
  };
  float*     h0   = (float*)alloc((size_t)kB * kT * kD * 4);
  float*     hA   = (float*)alloc((size_t)kB * kT * kD * 4);
  _Float16*  a16  = (_Float16*)alloc((size_t)kB * kT * kD * 2);
  _Float16*  q16  = (_Float16*)alloc((size_t)kB * kT * kD * 2);
  _Float16*  k16  = (_Float16*)alloc((size_t)kB * kT * kD * 2);
  _Float16*  vT16 = (_Float16*)alloc((size_t)kB * kT * kD * 2);
  _Float16*  c16  = (_Float16*)alloc((size_t)kB * kT * kD * 2);
  _Float16*  f16  = (_Float16*)alloc((size_t)kB * kT * kFF * 2);
  _Float16*  WqT  = (_Float16*)alloc((size_t)kN * kD * kD * 2);
  _Float16*  WkT  = (_Float16*)alloc((size_t)kN * kD * kD * 2);
  _Float16*  WvT  = (_Float16*)alloc((size_t)kN * kD * kD * 2);
  _Float16*  WoT  = (_Float16*)alloc((size_t)kN * kD * kD * 2);
  _Float16*  W1T  = (_Float16*)alloc((size_t)kN * kFF * kD * 2);
  _Float16*  W2T  = (_Float16*)alloc((size_t)kN * kD * kFF * 2);

  // 1) weights -> transposed f16
  {
    int tot = kN * kD * kD;
    int gb = (tot + 255) / 256;
    cvt_transpose_kernel<<<gb, 256, 0, stream>>>(Wq, WqT, kD, kD, tot);
    cvt_transpose_kernel<<<gb, 256, 0, stream>>>(Wk, WkT, kD, kD, tot);
    cvt_transpose_kernel<<<gb, 256, 0, stream>>>(Wv, WvT, kD, kD, tot);
    cvt_transpose_kernel<<<gb, 256, 0, stream>>>(Wo, WoT, kD, kD, tot);
    int tot2 = kN * kD * kFF;
    int gb2 = (tot2 + 255) / 256;
    cvt_transpose_kernel<<<gb2, 256, 0, stream>>>(W1, W1T, kD, kFF, tot2);
    cvt_transpose_kernel<<<gb2, 256, 0, stream>>>(W2, W2T, kFF, kD, tot2);
  }

  // 2) h0 = x*sqrt(D) + PE
  prep_kernel<<<(kB * kT * kD) / 256, 256, 0, stream>>>(x, h0);

  // 3) hn = LN(h0; g1,be1) -> f16
  ln_kernel<true><<<kB * kT, 256, 0, stream>>>(h0, g1, be1, lids, a16);

  // 4) Q/K/V projections (V stored transposed per head)
  dim3 gD(kT / 64, kD / 64, kB);
  gemm_kernel<STORE_F16><<<gD, 128, 0, stream>>>(a16, WqT, bq, nullptr, q16,
                                                 lids, kD, kD);
  gemm_kernel<STORE_F16><<<gD, 128, 0, stream>>>(a16, WkT, bk, nullptr, k16,
                                                 lids, kD, kD);
  gemm_kernel<STORE_VT><<<gD, 128, 0, stream>>>(a16, WvT, bv, nullptr, vT16,
                                                lids, kD, kD);

  // 5) attention
  dim3 gA(kT / 16, kH, kB);
  attn_kernel<<<gA, 32, 0, stream>>>(q16, k16, vT16, c16, xlen);

  // 6) h = h0 + ctx @ Wo + bo  (f32)
  gemm_kernel<STORE_F32RES><<<gD, 128, 0, stream>>>(c16, WoT, bo, h0, hA, lids,
                                                    kD, kD);

  // 7) hn2 = LN(h; g2,be2) -> f16
  ln_kernel<true><<<kB * kT, 256, 0, stream>>>(hA, g2, be2, lids, a16);

  // 8) ff = relu(hn2 @ W1 + b1) -> f16
  dim3 gF(kT / 64, kFF / 64, kB);
  gemm_kernel<STORE_F16RELU><<<gF, 128, 0, stream>>>(a16, W1T, b1, nullptr, f16,
                                                     lids, kD, kFF);

  // 9) h2 = h + ff @ W2 + b2 (f32, reuse h0 buffer)
  gemm_kernel<STORE_F32RES><<<gD, 128, 0, stream>>>(f16, W2T, b2, hA, h0, lids,
                                                    kFF, kD);

  // 10) out = LN(h2; gf,bef) -> f32 (one-hot selects exactly this branch)
  ln_kernel<false><<<kB * kT, 256, 0, stream>>>(h0, gf, bef, lids, d_out);
}
